// SG3MessageSynthesisLayer_64690797413094
// MI455X (gfx1250) — compile-verified, matrix operand order checked by reading, not run on hardware
//
#include <hip/hip_runtime.h>
#include <hip/hip_bf16.h>

// ---------------------------------------------------------------------------
// Types for CDNA5 WMMA (wave32, 16x16x32 bf16 -> f32)
// ---------------------------------------------------------------------------
typedef __attribute__((ext_vector_type(16))) __bf16 v16bf;
typedef __attribute__((ext_vector_type(8)))  __bf16 v8bf;
typedef __attribute__((ext_vector_type(8)))  float  v8f;

// Explicit LDS (addrspace 3) vector type so fragment reads are guaranteed to
// lower to ds_load_b128 (round-2 showed generic pointers through a runtime
// select got mis-lowered to global loads).
typedef __attribute__((address_space(3))) v8bf lds_v8bf;

// Load a 16-half WMMA fragment for one lane given an LDS byte address of the
// first 16B chunk. Per the CDNA5 ISA 16-bit A layout: lanes 0-15 need
// K {0..7, 16..23}, lanes 16-31 need K {8..15, 24..31} -> chunks at +0, +32B.
__device__ __forceinline__ v16bf ld_frag(unsigned addr) {
  v8bf lo = *(const lds_v8bf*)addr;
  v8bf hi = *(const lds_v8bf*)(addr + 32u);
  return __builtin_shufflevector(lo, hi, 0, 1, 2, 3, 4, 5, 6, 7,
                                 8, 9, 10, 11, 12, 13, 14, 15);
}

// CDNA5 async memory->LDS staging (ASYNCcnt-tracked). VDST holds the
// wave-relative LDS byte address, VADDR the 64-bit global address.
__device__ __forceinline__ void async_b128(unsigned int lds, const void* g) {
  asm volatile("global_load_async_to_lds_b128 %0, %1, off"
               :: "v"(lds), "v"(g) : "memory");
}
__device__ __forceinline__ void wait_async0() {
  asm volatile("s_wait_asynccnt 0x0" ::: "memory");
}

// ---------------------------------------------------------------------------
// K0: styles + message modulation + per-sample style normalization + bias.
// grid=16 (b), block=512
// ---------------------------------------------------------------------------
__global__ __launch_bounds__(512) void k_styles(
    const float* __restrict__ w, const float* __restrict__ msg,
    const float* __restrict__ aw, const float* __restrict__ ab,
    const float* __restrict__ bias, const float* __restrict__ msw,
    const float* __restrict__ mbw, float* __restrict__ sn,
    float* __restrict__ bcoef) {
  int b = blockIdx.x;
  int i = threadIdx.x;  // 0..511, doubles as Cin (style) and Cout (bias) index
  const float* wrow = w + b * 512;
  const float* arow = aw + i * 512;
  float acc = ab[i];
  for (int j = 0; j < 512; ++j) acc += wrow[j] * arow[j];
  const float* mrow = msg + b * 64;
  float mAcc = 0.f, bAcc = 0.f;
  for (int j = 0; j < 64; ++j) {
    mAcc += mrow[j] * msw[j * 512 + i];
    bAcc += mrow[j] * mbw[j * 512 + i];
  }
  float s = acc + 0.01f * mAcc;
  bcoef[b * 512 + i] = bias[i] + 0.01f * bAcc;

  __shared__ float red[512];
  red[i] = s * s;
  __syncthreads();
  for (int st = 256; st > 0; st >>= 1) {
    if (i < st) red[i] += red[i + st];
    __syncthreads();
  }
  float mean = red[0] * (1.0f / 512.0f);
  sn[b * 512 + i] = s * rsqrtf(mean);
}

// ---------------------------------------------------------------------------
// K1: weight normalization; wbt[t][o][i] bf16 (tap-major, N-major for B
// tiles); wn2sum[o][i] = sum_t wn^2.   grid=512 (o), block=256
// ---------------------------------------------------------------------------
__global__ __launch_bounds__(256) void k_wprep(
    const float* __restrict__ weight, __bf16* __restrict__ wbt,
    float* __restrict__ wn2sum) {
  int o = blockIdx.x;
  int tid = threadIdx.x;
  const float* wrow = weight + (size_t)o * 4608;  // [i*9 + t]
  float ss = 0.f;
  for (int idx = tid; idx < 4608; idx += 256) {
    float v = wrow[idx];
    ss += v * v;
  }
  __shared__ float red[256];
  red[tid] = ss;
  __syncthreads();
  for (int st = 128; st > 0; st >>= 1) {
    if (tid < st) red[tid] += red[tid + st];
    __syncthreads();
  }
  float rs = rsqrtf(red[0] * (1.0f / 4608.0f));
  for (int idx = tid; idx < 4608; idx += 256) {
    int i = idx / 9, t = idx % 9;
    wbt[((size_t)t * 512 + o) * 512 + i] = (__bf16)(wrow[idx] * rs);
  }
  for (int i = tid; i < 512; i += 256) {
    float s2 = 0.f;
    for (int t = 0; t < 9; ++t) {
      float v = wrow[i * 9 + t] * rs;
      s2 += v * v;
    }
    wn2sum[(size_t)o * 512 + i] = s2;
  }
}

// ---------------------------------------------------------------------------
// K2a: zero-fill the 2-pixel pad border of the padded HWC input.
// grid=(1600, 16), block=512
// ---------------------------------------------------------------------------
__global__ __launch_bounds__(512) void k_xzero(__bf16* __restrict__ xs) {
  int pos = blockIdx.x;  // 0..1599
  int b = blockIdx.y;
  int i = threadIdx.x;
  int h = pos / 40, wq = pos % 40;
  if (h >= 2 && h < 38 && wq >= 2 && wq < 38) return;
  xs[((size_t)(b * 1600 + pos)) * 512 + i] = (__bf16)0.f;
}

// ---------------------------------------------------------------------------
// K2b: LDS-tiled NCHW->padded-NHWC transpose with style folded in.
//   Reads contiguous 36-float channel rows, transposes in LDS, writes 128B
//   channel-major runs.  grid=(8 ch-chunks, 36 rows, 16 b), block=256
// ---------------------------------------------------------------------------
__global__ __launch_bounds__(256) void k_xprep(
    const float* __restrict__ x, const float* __restrict__ sn,
    __bf16* __restrict__ xs) {
  int c0 = blockIdx.x * 64;
  int h = blockIdx.y;  // 0..35
  int b = blockIdx.z;
  int tid = threadIdx.x;
  __shared__ float tile[36 * 64];  // [w][ch]
  for (int idx = tid; idx < 64 * 36; idx += 256) {
    int ch = idx / 36, w = idx % 36;
    float v = x[(((size_t)(b * 512 + c0 + ch)) * 36 + h) * 36 + w] *
              sn[b * 512 + c0 + ch];
    tile[w * 64 + ch] = v;
  }
  __syncthreads();
  __bf16* dst = xs + ((size_t)(b * 1600 + (h + 2) * 40 + 2)) * 512 + c0;
  for (int idx = tid; idx < 64 * 36; idx += 256) {
    int w = idx >> 6, ch = idx & 63;
    dst[(size_t)w * 512 + ch] = (__bf16)tile[idx];
  }
}

// ---------------------------------------------------------------------------
// K3: demodulation scale:  dg[b,o] = rsqrt(sum_i sn^2 * wn2sum[o,i] + 1e-8)*ig
// grid=16 (b), block=512 (o)
// ---------------------------------------------------------------------------
__global__ __launch_bounds__(512) void k_dscale(
    const float* __restrict__ sn, const float* __restrict__ wn2sum,
    const float* __restrict__ ema, float* __restrict__ dg) {
  int b = blockIdx.x;
  int o = threadIdx.x;
  float ig = rsqrtf(ema[0]);
  __shared__ float sn2[512];
  float sv = sn[b * 512 + o];
  sn2[o] = sv * sv;
  __syncthreads();
  const float* row = wn2sum + (size_t)o * 512;
  float acc = 0.f;
  for (int i = 0; i < 512; ++i) acc += sn2[i] * row[i];
  dg[b * 512 + o] = rsqrtf(acc + 1e-8f) * ig;
}

// ---------------------------------------------------------------------------
// K4: implicit-GEMM conv via v_wmma_f32_16x16x32_bf16.
//   Block tile: 64 positions x 128 channels; 8 waves as 2(pos) x 4(chan);
//   each wave owns a 32x32 tile = 4 accumulators -> 4 WMMAs per k-step.
//   144 k-steps (9 taps x 16 cin-chunks of K=32).
//   Double-buffered LDS fed by GLOBAL_LOAD_ASYNC_TO_LDS_B128 (ASYNCcnt),
//   single barrier per k-step; all LDS addressing is explicit addrspace(3).
//   Epilogue: y1[b,o,p] = acc * dg[b,o] + bcoef[b,o]   (p over 38*38=1444)
// grid=(4 o-tiles, 23 p-tiles, 16 b), block=256
// ---------------------------------------------------------------------------
__global__ __launch_bounds__(256) void k_conv(
    const __bf16* __restrict__ xs, const __bf16* __restrict__ wbt,
    const float* __restrict__ dg, const float* __restrict__ bcoef,
    float* __restrict__ y1) {
  int o0 = blockIdx.x * 128;
  int p0 = blockIdx.y * 64;
  int b = blockIdx.z;
  int tid = threadIdx.x;
  int lane = tid & 31, wave = tid >> 5;
  int wy = wave >> 2;  // position half (0,1): rows wy*32 .. wy*32+31
  int wx = wave & 3;   // channel quarter: cols wx*32 .. wx*32+31

  // LDS: [A buf0 4K][A buf1 4K][B buf0 8K][B buf1 8K], addressed by byte off.
  __shared__ __attribute__((aligned(16))) unsigned char smem[24 * 1024];
  unsigned sbase = (unsigned)(uintptr_t)smem;  // wave-relative LDS byte addr
  unsigned A0 = sbase, A1 = sbase + 4096u;
  unsigned B0 = sbase + 8192u, B1 = sbase + 16384u;

  // Staging geometry: 256 threads; A = 64 rows x 4x16B, B = 128 rows x 4x16B
  // (two B rows per thread).
  int arow = tid >> 2;      // 0..63 (A row; also B rows arow and arow+64)
  int col = (tid & 3) * 8;  // halves (16B chunk)
  int pa = p0 + arow;
  if (pa > 1443) pa = 1443;
  int aoy = pa / 38, aox = pa % 38;
  size_t abase = ((size_t)(b * 1600 + aoy * 40 + aox)) * 512 + col;
  size_t bbase0 = ((size_t)(o0 + arow)) * 512 + col;
  size_t bbase1 = bbase0 + (size_t)64 * 512;
  unsigned dA = (unsigned)((arow * 32 + col) * 2);
  unsigned dB1 = dA + 64u * 64u;  // row arow+64 within B buffer

  auto issue = [&](int step, unsigned Ab, unsigned Bb) {
    int t = step >> 4;          // tap 0..8
    int cc = (step & 15) << 5;  // cin chunk offset (elements)
    int ky = t / 3, kx = t % 3;
    const __bf16* ga = xs + abase + (size_t)((ky * 40 + kx) * 512 + cc);
    const __bf16* gb = wbt + ((size_t)t << 18) + cc;  // + t*512*512
    async_b128(Ab + dA, ga);
    async_b128(Bb + dA, gb + bbase0);
    async_b128(Bb + dB1, gb + bbase1);
  };

  // Per-lane fragment byte offsets within a buffer (first 16B chunk).
  unsigned kb2 = (lane < 16) ? 0u : 16u;
  int fr = lane & 15;
  unsigned ar0 = (unsigned)(((wy * 32 + fr) * 32) * 2) + kb2;
  unsigned ar1 = ar0 + 1024u;  // +16 rows
  unsigned br0 = (unsigned)(((wx * 32 + fr) * 32) * 2) + kb2;
  unsigned br1 = br0 + 1024u;

  v8f acc00 = {}, acc01 = {}, acc10 = {}, acc11 = {};

  issue(0, A0, B0);
  for (int step = 0; step < 144; step += 2) {
    // ---- phase 0: consume buf0, prefetch into buf1 ----
    wait_async0();
    __syncthreads();
    if (step + 1 < 144) issue(step + 1, A1, B1);
    {
      v16bf a0 = ld_frag(A0 + ar0);
      v16bf a1 = ld_frag(A0 + ar1);
      v16bf b0f = ld_frag(B0 + br0);
      v16bf b1f = ld_frag(B0 + br1);
      acc00 = __builtin_amdgcn_wmma_f32_16x16x32_bf16(false, a0, false, b0f,
                                                      (short)0, acc00, false, false);
      acc01 = __builtin_amdgcn_wmma_f32_16x16x32_bf16(false, a0, false, b1f,
                                                      (short)0, acc01, false, false);
      acc10 = __builtin_amdgcn_wmma_f32_16x16x32_bf16(false, a1, false, b0f,
                                                      (short)0, acc10, false, false);
      acc11 = __builtin_amdgcn_wmma_f32_16x16x32_bf16(false, a1, false, b1f,
                                                      (short)0, acc11, false, false);
    }
    // ---- phase 1: consume buf1, prefetch into buf0 ----
    wait_async0();
    __syncthreads();
    if (step + 2 < 144) issue(step + 2, A0, B0);
    {
      v16bf a0 = ld_frag(A1 + ar0);
      v16bf a1 = ld_frag(A1 + ar1);
      v16bf b0f = ld_frag(B1 + br0);
      v16bf b1f = ld_frag(B1 + br1);
      acc00 = __builtin_amdgcn_wmma_f32_16x16x32_bf16(false, a0, false, b0f,
                                                      (short)0, acc00, false, false);
      acc01 = __builtin_amdgcn_wmma_f32_16x16x32_bf16(false, a0, false, b1f,
                                                      (short)0, acc01, false, false);
      acc10 = __builtin_amdgcn_wmma_f32_16x16x32_bf16(false, a1, false, b0f,
                                                      (short)0, acc10, false, false);
      acc11 = __builtin_amdgcn_wmma_f32_16x16x32_bf16(false, a1, false, b1f,
                                                      (short)0, acc11, false, false);
    }
  }

  // Epilogue. 16x16 f32 C/D layout: VGPR r, lanes 0-15 -> M=r, lanes 16-31 ->
  // M=r+8; N = lane%16.
  int mb = (lane < 16) ? 0 : 8;
#pragma unroll
  for (int mi = 0; mi < 2; ++mi) {
#pragma unroll
    for (int ni = 0; ni < 2; ++ni) {
      const v8f& a = (mi == 0) ? (ni == 0 ? acc00 : acc01)
                               : (ni == 0 ? acc10 : acc11);
      int o = o0 + wx * 32 + ni * 16 + fr;
      float dv = dg[b * 512 + o];
      float bv = bcoef[b * 512 + o];
      float* dst = y1 + ((size_t)(b * 512 + o)) * 1444;
#pragma unroll
      for (int r = 0; r < 8; ++r) {
        int p = p0 + wy * 32 + mi * 16 + mb + r;
        if (p < 1444) dst[p] = a[r] * dv + bv;
      }
    }
  }
}

// ---------------------------------------------------------------------------
// K5: fused filtered_lrelu per (b,c) image, all intermediates in LDS.
//   38x38 -> up-FIR(V) 82x38 -> up-FIR(H)+gain4+lrelu*sqrt2+clamp 82x82
//         -> down-FIR(V, even rows) 36x82 -> down-FIR(H, even cols) 36x36
// grid=8192 (b*512+c), block=256
// ---------------------------------------------------------------------------
__global__ __launch_bounds__(256) void k_flrelu(
    const float* __restrict__ y1, const float* __restrict__ fU,
    const float* __restrict__ fD, float* __restrict__ out) {
  int bc = blockIdx.x;
  int tid = threadIdx.x;
  __shared__ float bin[38 * 38];
  __shared__ float bv[82 * 38];
  __shared__ float bu[82 * 82];
  __shared__ float bdv[36 * 82];

  float fu[12], fd[12];
#pragma unroll
  for (int j = 0; j < 12; ++j) {
    fu[j] = fU[11 - j];  // reference flips the filter before correlation
    fd[j] = fD[11 - j];
  }

  const float* src = y1 + (size_t)bc * 1444;
  for (int idx = tid; idx < 1444; idx += 256) bin[idx] = src[idx];
  __syncthreads();

  // vertical up: zero-insert x2, pad 9 each side, 12-tap
  for (int idx = tid; idx < 82 * 38; idx += 256) {
    int r = idx / 38, c = idx % 38;
    float a = 0.f;
#pragma unroll
    for (int j = 0; j < 12; ++j) {
      int t = r + j - 9;
      if (t >= 0 && t <= 74 && !(t & 1)) a += fu[j] * bin[(t >> 1) * 38 + c];
    }
    bv[idx] = a;
  }
  __syncthreads();

  // horizontal up + gain 4 + leaky-relu * sqrt(2) + clamp +-256
  for (int idx = tid; idx < 82 * 82; idx += 256) {
    int r = idx / 82, c = idx % 82;
    float a = 0.f;
#pragma unroll
    for (int j = 0; j < 12; ++j) {
      int t = c + j - 9;
      if (t >= 0 && t <= 74 && !(t & 1)) a += fu[j] * bv[r * 38 + (t >> 1)];
    }
    a *= 4.0f;
    float v = (a >= 0.f ? a : 0.2f * a) * 1.41421356237309504880f;
    v = fminf(fmaxf(v, -256.0f), 256.0f);
    bu[idx] = v;
  }
  __syncthreads();

  // vertical down (keep even rows only): r = 2*ri + j
  for (int idx = tid; idx < 36 * 82; idx += 256) {
    int ri = idx / 82, c = idx % 82;
    float a = 0.f;
#pragma unroll
    for (int j = 0; j < 12; ++j) a += fd[j] * bu[(2 * ri + j) * 82 + c];
    bdv[idx] = a;
  }
  __syncthreads();

  // horizontal down (keep even cols only) + store
  float* dst = out + (size_t)bc * 1296;
  for (int idx = tid; idx < 1296; idx += 256) {
    int ri = idx / 36, ci = idx % 36;
    float a = 0.f;
#pragma unroll
    for (int j = 0; j < 12; ++j) a += fd[j] * bdv[ri * 82 + 2 * ci + j];
    dst[idx] = a;
  }
}

// ---------------------------------------------------------------------------
// Host launcher
// ---------------------------------------------------------------------------
extern "C" void kernel_launch(void* const* d_in, const int* in_sizes, int n_in,
                              void* d_out, int out_size, void* d_ws,
                              size_t ws_size, hipStream_t stream) {
  (void)in_sizes; (void)n_in; (void)out_size; (void)ws_size;
  const float* x   = (const float*)d_in[0];   // [16,512,36,36]
  const float* w   = (const float*)d_in[1];   // [16,512]
  const float* msg = (const float*)d_in[2];   // [16,64]
  const float* aw  = (const float*)d_in[3];   // [512,512]
  const float* ab  = (const float*)d_in[4];   // [512]
  const float* wt  = (const float*)d_in[5];   // [512,512,3,3]
  const float* bias= (const float*)d_in[6];   // [512]
  const float* msw = (const float*)d_in[7];   // [64,512]
  const float* mbw = (const float*)d_in[8];   // [64,512]
  const float* fU  = (const float*)d_in[9];   // [12]
  const float* fD  = (const float*)d_in[10];  // [12]
  const float* ema = (const float*)d_in[11];  // scalar
  float* out = (float*)d_out;                 // [16,512,36,36]

  // Workspace layout (256B aligned)
  char* ws = (char*)d_ws;
  size_t off = 0;
  auto alloc = [&](size_t bytes) {
    char* p = ws + off;
    off += (bytes + 255) & ~(size_t)255;
    return p;
  };
  float*  sn     = (float*)alloc(16 * 512 * sizeof(float));
  float*  bcoef  = (float*)alloc(16 * 512 * sizeof(float));
  float*  dg     = (float*)alloc(16 * 512 * sizeof(float));
  float*  wn2sum = (float*)alloc((size_t)512 * 512 * sizeof(float));
  __bf16* wbt    = (__bf16*)alloc((size_t)9 * 512 * 512 * sizeof(__bf16));
  __bf16* xs     = (__bf16*)alloc((size_t)16 * 1600 * 512 * sizeof(__bf16));
  float*  y1     = (float*)alloc((size_t)16 * 512 * 1444 * sizeof(float));

  k_styles<<<16, 512, 0, stream>>>(w, msg, aw, ab, bias, msw, mbw, sn, bcoef);
  k_wprep<<<512, 256, 0, stream>>>(wt, wbt, wn2sum);
  k_xzero<<<dim3(1600, 16), 512, 0, stream>>>(xs);
  k_xprep<<<dim3(8, 36, 16), 256, 0, stream>>>(x, sn, xs);
  k_dscale<<<16, 512, 0, stream>>>(sn, wn2sum, ema, dg);
  k_conv<<<dim3(4, 23, 16), 256, 0, stream>>>(xs, wbt, dg, bcoef, y1);
  k_flrelu<<<8192, 256, 0, stream>>>(y1, fU, fD, out);
}
